// StemBlock_5179730559087
// MI455X (gfx1250) — matrix operand order repaired and checked by reading, not compiled, not run on hardware
//
#include <hip/hip_runtime.h>

typedef __attribute__((ext_vector_type(8)))  _Float16     v8h;
typedef __attribute__((ext_vector_type(16))) _Float16     v16h;
typedef __attribute__((ext_vector_type(8)))  float        v8f;
typedef __attribute__((ext_vector_type(4)))  unsigned int v4u;

#define B_     64
#define CIN    3
#define HIN    224
#define WIN    224
#define COUT   64
#define KS     7
#define STRIDE 2
#define PAD    3
#define HOUT   112
#define WOUT   112
#define KRAW   147            // 3*7*7
#define KP     160            // K padded to 5 x 32
#define NT     128            // output pixels per block
#define SA     168            // LDS row stride (halfs), multiple of 8, bank-spread
#define SB     168
#define NPIX   (B_*HOUT*WOUT) // 802816
#define NBLK   (NPIX/NT)      // 6272 exact

union Frag16 { v16h v; v8h h[2]; };

// ---------- kernel 1: per-channel fake-quant + BN/bias folding ----------
__global__ __launch_bounds__(160)
void stem_prep(const float* __restrict__ weight,
               const float* __restrict__ gamma_, const float* __restrict__ beta_,
               const float* __restrict__ mean_,  const float* __restrict__ var_,
               const float* __restrict__ bias1_,
               _Float16* __restrict__ wq, float* __restrict__ c0, float* __restrict__ c1)
{
    __shared__ float red[160];
    const int c = blockIdx.x, t = threadIdx.x;
    float w = 0.f;
    if (t < KRAW) {
        int ci = t / 49, r = t % 49, kh = r / 7, kw = r % 7;
        w = weight[((c*CIN + ci)*KS + kh)*KS + kw];
    }
    red[t] = fabsf(w);
    __syncthreads();
    for (int s = 128; s > 0; s >>= 1) {
        if (t < s && t + s < 160) red[t] = fmaxf(red[t], red[t+s]);
        __syncthreads();
    }
    float scale = fmaxf(red[0] * (1.f/127.f), 1e-8f);
    // round-half-to-even to match jnp.round; integers <=127 are exact in f16
    float q = (t < KRAW) ? fminf(fmaxf(rintf(w/scale), -127.f), 127.f) : 0.f;
    wq[c*KP + t] = (_Float16)q;
    if (t == 0) {
        float inv = rsqrtf(var_[c] + 1e-5f);
        float g   = gamma_[c] * inv;
        c0[c] = scale * g;                               // multiplies raw q-accumulator
        c1[c] = beta_[c] + bias1_[c] - mean_[c] * g;     // additive fold (pre-PReLU)
    }
}

// ---------- kernel 2: implicit GEMM conv via v_wmma_f32_16x16x32_f16 ----------
__global__ __launch_bounds__(256)
void stem_conv(const float* __restrict__ x,
               const _Float16* __restrict__ wq,
               const float* __restrict__ c0, const float* __restrict__ c1,
               const float* __restrict__ pa, const float* __restrict__ pb2,
               float* __restrict__ out)
{
    __shared__ _Float16 As[COUT * SA];   // 21504 B
    __shared__ _Float16 Bs[NT   * SB];   // 43008 B
    const int t  = threadIdx.x;
    const int p0 = blockIdx.x * NT;

    // --- load A (64 x 160 halfs = 1280 x 16B), restride 160 -> 168 ---
    {
        const v4u* src = (const v4u*)wq;
#pragma unroll
        for (int i = 0; i < 5; ++i) {
            int idx = t + i * 256;          // 0..1279
            int row = idx / 20, col = idx % 20;
            *(v4u*)&As[row*SA + col*8] = src[idx];
        }
    }
    // --- im2col B: 2 threads per output pixel, 80 K-values each ---
    {
        int n = t >> 1, sub = t & 1;
        int p = p0 + n;
        int ow = p % WOUT; int tmp = p / WOUT; int oh = tmp % HOUT; int b = tmp / HOUT;
        int ih0 = oh*STRIDE - PAD, iw0 = ow*STRIDE - PAD;
        const float* xb = x + (size_t)b * (CIN*HIN*WIN);
        for (int kk = 0; kk < 80; ++kk) {
            int k = sub*80 + kk;
            float v = 0.f;
            if (k < KRAW) {
                int ci = k / 49, r = k % 49, kh = r / 7, kw = r % 7;
                int ih = ih0 + kh, iw = iw0 + kw;
                if (ih >= 0 && ih < HIN && iw >= 0 && iw < WIN)
                    v = xb[(ci*HIN + ih)*WIN + iw];
            }
            Bs[n*SB + k] = (_Float16)v;
        }
    }
    __syncthreads();

    const int lane = t & 31, wave = t >> 5;
    const int mt   = wave & 3;       // M tile (16 rows)
    const int ng   = wave >> 2;      // N group (64 cols)
    const int lrow = lane & 15;
    const int hi16 = lane >> 4;

    v8f acc[4] = {};                 // 4 N-subtiles of 16
    const _Float16* arow = &As[(mt*16 + lrow) * SA];
#pragma unroll
    for (int kb = 0; kb < KP; kb += 32) {
        // A fragment: lanes 0-15 take K halves {0..7,16..23}, lanes 16-31 {8..15,24..31}
        Frag16 a;
        int ka = kb + hi16*8;
        a.h[0] = *(const v8h*)&arow[ka];
        a.h[1] = *(const v8h*)&arow[ka + 16];
        // B fragment: lanes 0-15 take K halves {0..7,16..23}, lanes 16-31 {8..15,24..31}
        int kbb = kb + hi16*8;
#pragma unroll
        for (int nt = 0; nt < 4; ++nt) {
            const _Float16* brow = &Bs[(ng*64 + nt*16 + lrow) * SB + kbb];
            Frag16 bf;
            bf.h[0] = *(const v8h*)&brow[0];
            bf.h[1] = *(const v8h*)&brow[16];
            acc[nt] = __builtin_amdgcn_wmma_f32_16x16x32_f16(
                false, a.v, false, bf.v, (short)0, acc[nt], false, false);
        }
    }

    // --- fused epilogue: affine(BN+bias1) -> PReLU -> +bias2, scatter to NCHW ---
    const int mbase = mt*16 + hi16*8;      // C/D layout: VGPR r -> M = mbase + r
    float f0[8], f1[8], fa[8], fb[8];
#pragma unroll
    for (int r = 0; r < 8; ++r) {
        int m = mbase + r;
        f0[r] = c0[m]; f1[r] = c1[m]; fa[r] = pa[m]; fb[r] = pb2[m];
    }
#pragma unroll
    for (int nt = 0; nt < 4; ++nt) {
        int p = p0 + ng*64 + nt*16 + lrow;
        int ow = p % WOUT; int tmp = p / WOUT; int oh = tmp % HOUT; int b = tmp / HOUT;
        float* op = out + (((size_t)b * COUT) * HOUT + oh) * WOUT + ow;
#pragma unroll
        for (int r = 0; r < 8; ++r) {
            float v = acc[nt][r] * f0[r] + f1[r];
            v = v > 0.f ? v : v * fa[r];
            v += fb[r];
            op[(size_t)(mbase + r) * (HOUT*WOUT)] = v;
        }
    }
}

extern "C" void kernel_launch(void* const* d_in, const int* in_sizes, int n_in,
                              void* d_out, int out_size, void* d_ws, size_t ws_size,
                              hipStream_t stream) {
    (void)in_sizes; (void)n_in; (void)out_size; (void)ws_size;
    const float* x      = (const float*)d_in[0];
    const float* weight = (const float*)d_in[1];
    const float* gamma_ = (const float*)d_in[2];
    const float* beta_  = (const float*)d_in[3];
    const float* mean_  = (const float*)d_in[4];
    const float* var_   = (const float*)d_in[5];
    const float* bias1_ = (const float*)d_in[6];
    const float* prelu  = (const float*)d_in[7];
    const float* bias2  = (const float*)d_in[8];
    float* out = (float*)d_out;

    _Float16* wq = (_Float16*)d_ws;
    float* c0 = (float*)((char*)d_ws + (size_t)COUT * KP * sizeof(_Float16));
    float* c1 = c0 + COUT;

    stem_prep<<<COUT, 160, 0, stream>>>(weight, gamma_, beta_, mean_, var_, bias1_,
                                        wq, c0, c1);
    stem_conv<<<NBLK, 256, 0, stream>>>(x, wq, c0, c1, prelu, bias2, out);
}
